// GraphEncoder_43559558316699
// MI455X (gfx1250) — matrix operand order repaired and checked by reading, not compiled, not run on hardware
//
#include <hip/hip_runtime.h>
#include <hip/hip_bf16.h>
#include <stdint.h>

typedef __attribute__((ext_vector_type(16))) __bf16 v16bf;
typedef __attribute__((ext_vector_type(8)))  float  v8f;

#define FDIM   128          // feature dim for every matrix
#define KDIM   256          // concat K = [agg | root]
#define LPITCH 264          // padded LDS row pitch in bf16 elements (528 B -> 4-bank skew)

// ---------------------------------------------------------------------------
// f32 -> bf16 (round to nearest even)
__device__ __forceinline__ uint16_t f32_to_bf16(float f) {
    union { float f; uint32_t u; } v; v.f = f;
    uint32_t u = v.u + 0x7FFFu + ((v.u >> 16) & 1u);
    return (uint16_t)(u >> 16);
}

// ---------------------------------------------------------------------------
// zero a float buffer (grid-stride, float4)
__global__ void zero_kernel(float* __restrict__ p, long nElem) {
    long i4 = ((long)blockIdx.x * blockDim.x + threadIdx.x) * 4;
    long stride = (long)gridDim.x * blockDim.x * 4;
    for (; i4 + 3 < nElem; i4 += stride) {
        *(float4*)(p + i4) = make_float4(0.f, 0.f, 0.f, 0.f);
    }
    if (i4 < nElem) { for (long j = i4; j < nElem && j < i4 + 4; ++j) p[j] = 0.f; }
}

// ---------------------------------------------------------------------------
// degree: one thread per edge
__global__ void degree_kernel(const int64_t* __restrict__ ei, long E,
                              float* __restrict__ deg) {
    long e = (long)blockIdx.x * blockDim.x + threadIdx.x;
    if (e >= E) return;
    long d = ei[E + e];
    atomicAdd(deg + d, 1.0f);
}

// ---------------------------------------------------------------------------
// scatter-add features: one wave (32 lanes) per edge, 4 floats per lane
__global__ void scatter_kernel(const float* __restrict__ feat,
                               const int64_t* __restrict__ ei, long E,
                               float* __restrict__ agg) {
    long gid  = (long)blockIdx.x * blockDim.x + threadIdx.x;
    long eid  = gid >> 5;
    int  lane = (int)(gid & 31);
    if (eid >= E) return;
    // speculative prefetch of upcoming edge indices (global_prefetch_b8)
    __builtin_prefetch(ei + eid + 4096, 0, 0);
    long s = ei[eid];
    long d = ei[E + eid];
    const float4 v = *(const float4*)(feat + s * FDIM + lane * 4);
    float* dst = agg + d * FDIM + lane * 4;
    atomicAdd(dst + 0, v.x);
    atomicAdd(dst + 1, v.y);
    atomicAdd(dst + 2, v.z);
    atomicAdd(dst + 3, v.w);
}

// ---------------------------------------------------------------------------
// Fused SAGE layer:  out = act( [agg*deg^-1 | root] @ [Wl ; Wr] + b )
// block = 256 threads (8 waves), 128 rows per block, full 128 output cols.
// A and Wt staged in LDS as bf16; v_wmma_f32_16x16x32_bf16 main loop.
__global__ void __launch_bounds__(256, 2)
sage_gemm_kernel(const float* __restrict__ agg,
                 const float* __restrict__ deg,
                 const float* __restrict__ root,
                 const float* __restrict__ Wl,
                 const float* __restrict__ Wr,
                 const float* __restrict__ bias,
                 float* __restrict__ out,
                 long n, int relu) {
    extern __shared__ char smem[];
    uint16_t* A_lds = (uint16_t*)smem;                         // 128 x LPITCH bf16
    uint16_t* W_lds = A_lds + 128 * LPITCH;                    // 128 x LPITCH bf16 (N-major, K contig)
    float*    b_lds = (float*)(W_lds + 128 * LPITCH);          // 128 f32

    const int  tid     = threadIdx.x;
    const long rowBase = (long)blockIdx.x * 128;

    // ---- stage A = [agg/deg | root] as bf16, 128 rows x 256 cols -----------
    for (int i = tid; i < 128 * 64; i += 256) {          // 64 float4 groups / row
        int  row  = i >> 6;
        int  col  = (i & 63) * 4;
        long rg   = rowBase + row;
        float4 v  = make_float4(0.f, 0.f, 0.f, 0.f);
        if (rg < n) {
            if (col < FDIM) {
                float sc = 1.0f / fmaxf(deg[rg], 1.0f);
                v = *(const float4*)(agg + rg * FDIM + col);
                v.x *= sc; v.y *= sc; v.z *= sc; v.w *= sc;
            } else {
                v = *(const float4*)(root + rg * FDIM + (col - FDIM));
            }
        }
        uint16_t* p = A_lds + row * LPITCH + col;
        uint2 pk;
        pk.x = (uint32_t)f32_to_bf16(v.x) | ((uint32_t)f32_to_bf16(v.y) << 16);
        pk.y = (uint32_t)f32_to_bf16(v.z) | ((uint32_t)f32_to_bf16(v.w) << 16);
        *(uint2*)p = pk;
    }

    // ---- stage weights transposed: W_lds[nCol][k], k<128 -> Wl, else Wr ----
    for (int i = tid; i < KDIM * FDIM; i += 256) {
        int k = i >> 7;           // 0..255
        int nc = i & 127;         // coalesced over n in global reads
        float w = (k < FDIM) ? Wl[k * FDIM + nc] : Wr[(k - FDIM) * FDIM + nc];
        W_lds[nc * LPITCH + k] = f32_to_bf16(w);
    }
    if (tid < FDIM) b_lds[tid] = bias[tid];
    __syncthreads();

    // ---- WMMA main loop ----------------------------------------------------
    const int wid  = tid >> 5;          // wave -> 16-row slab
    const int lane = tid & 31;
    const int ll   = lane & 15;
    const int hi   = lane >> 4;

    const uint16_t* Arow = A_lds + (wid * 16 + ll) * LPITCH;

    v8f acc[8] = {};                    // 8 column tiles of 16x16 f32

    union V16 { v16bf v; uint4 q[2]; };

    #pragma unroll
    for (int k0 = 0; k0 < KDIM; k0 += 32) {
        // A lane layout (16-bit A 16x32): lanes 0-15 hold K {0-7,16-23},
        // lanes 16-31 hold K {8-15,24-31}, M = lane&15.
        V16 ua;
        const int ka = k0 + (hi ? 8 : 0);
        ua.q[0] = *(const uint4*)(Arow + ka);
        ua.q[1] = *(const uint4*)(Arow + ka + 16);
        #pragma unroll
        for (int t = 0; t < 8; ++t) {
            // B lane layout (16-bit B 32x16): lanes 0-15 hold K {0-7,16-23},
            // lanes 16-31 hold K {8-15,24-31}, N = lane&15.
            const uint16_t* Brow = W_lds + (t * 16 + ll) * LPITCH;
            V16 ub;
            const int kb = k0 + hi * 8;
            ub.q[0] = *(const uint4*)(Brow + kb);
            ub.q[1] = *(const uint4*)(Brow + kb + 16);
            acc[t] = __builtin_amdgcn_wmma_f32_16x16x32_bf16(
                false, ua.v, false, ub.v, (short)0, acc[t], false, false);
        }
    }

    // ---- epilogue: bias, activation, store per C/D layout ------------------
    #pragma unroll
    for (int t = 0; t < 8; ++t) {
        const int col = t * 16 + ll;
        const float bv = b_lds[col];
        #pragma unroll
        for (int v = 0; v < 8; ++v) {
            const int m = v + hi * 8;               // C/D: vgpr v -> M=v (+8 upper half)
            const long rg = rowBase + wid * 16 + m;
            if (rg < n) {
                float val = acc[t][v] + bv;
                if (relu) val = fmaxf(val, 0.f);
                out[rg * FDIM + col] = val;
            }
        }
    }
}

// ---------------------------------------------------------------------------
extern "C" void kernel_launch(void* const* d_in, const int* in_sizes, int n_in,
                              void* d_out, int out_size, void* d_ws, size_t ws_size,
                              hipStream_t stream) {
    const float*   x   = (const float*)d_in[0];
    const int64_t* ei  = (const int64_t*)d_in[1];
    const float*   W1l = (const float*)d_in[2];
    const float*   b1l = (const float*)d_in[3];
    const float*   W1r = (const float*)d_in[4];
    const float*   W2l = (const float*)d_in[5];
    const float*   b2l = (const float*)d_in[6];
    const float*   W2r = (const float*)d_in[7];

    const long n = (long)in_sizes[0] / FDIM;
    const long E = (long)in_sizes[1] / 2;

    // workspace carve: agg[n*128] | deg[n (padded)] | h1[n*128]
    float* agg = (float*)d_ws;
    float* deg = agg + n * FDIM;
    float* h1  = deg + ((n + 31) & ~31L);

    float* outp = (float*)d_out;

    const int zeroBlocks = 2048;
    const long edgeThreads = E * 32;
    const int scatterBlocks = (int)((edgeThreads + 255) / 256);
    const int degBlocks = (int)((E + 255) / 256);
    const int gemmBlocks = (int)((n + 127) / 128);
    const size_t smemBytes = (size_t)(2 * 128 * LPITCH) * sizeof(uint16_t)
                           + FDIM * sizeof(float);

    // ---------------- layer 1 ----------------
    zero_kernel<<<zeroBlocks, 256, 0, stream>>>(agg, n * FDIM + ((n + 31) & ~31L));
    degree_kernel<<<degBlocks, 256, 0, stream>>>(ei, E, deg);
    scatter_kernel<<<scatterBlocks, 256, 0, stream>>>(x, ei, E, agg);
    sage_gemm_kernel<<<gemmBlocks, 256, smemBytes, stream>>>(
        agg, deg, x, W1l, W1r, b1l, h1, n, /*relu=*/1);

    // ---------------- layer 2 ----------------
    zero_kernel<<<zeroBlocks, 256, 0, stream>>>(agg, n * FDIM);
    scatter_kernel<<<scatterBlocks, 256, 0, stream>>>(h1, ei, E, agg);
    sage_gemm_kernel<<<gemmBlocks, 256, smemBytes, stream>>>(
        agg, deg, h1, W2l, W2r, b2l, outp, n, /*relu=*/0);
}
